// ContentContrastiveLoss_90658169684671
// MI455X (gfx1250) — matrix operand order repaired and checked by reading, not compiled, not run on hardware
//
#include <hip/hip_runtime.h>

typedef __attribute__((ext_vector_type(16))) __bf16 v16bf;
typedef __attribute__((ext_vector_type(8)))  float  v8f;
typedef __attribute__((ext_vector_type(4)))  float  v4f;
typedef __attribute__((ext_vector_type(4)))  int    v4i;

#define N_ROWS  2048
#define DIMK    8192
#define MARGINF 0.2f

#define BM 128
#define BN 128
#define BK 64
#define LDSS 72                 // padded row stride in ushorts (144B; 36-dword stride spreads banks)
#define KTILES (DIMK / BK)      // 128 iterations

#if __has_builtin(__builtin_amdgcn_global_load_async_to_lds_b128)
#define HAVE_ASYNC_LDS 1
#else
#define HAVE_ASYNC_LDS 0
#endif

#if HAVE_ASYNC_LDS
typedef __attribute__((address_space(1))) v4i gv4i;   // global int4
typedef __attribute__((address_space(3))) v4i lv4i;   // LDS int4
static __device__ __forceinline__ void async_cp_b128(const void* g, void* l) {
    // GLOBAL_LOAD_ASYNC_TO_LDS_B128: per-lane 16B global->LDS, tracked by ASYNCcnt
    __builtin_amdgcn_global_load_async_to_lds_b128((gv4i*)g, (lv4i*)l, 0, 0);
}
static __device__ __forceinline__ void wait_async0() {
#if __has_builtin(__builtin_amdgcn_s_wait_asynccnt)
    __builtin_amdgcn_s_wait_asynccnt(0);
#else
    asm volatile("s_wait_asynccnt 0x0" ::: "memory");
#endif
}
#endif

static __device__ __forceinline__ unsigned int f32_to_bf16(float f) {
    unsigned int u = __float_as_uint(f);
    u += 0x7FFFu + ((u >> 16) & 1u);   // round-to-nearest-even
    return u >> 16;
}

// ---------------------------------------------------------------------------
// Kernel 1: per-row L2 norms (fp32), exact diagonal dot, pack bf16-normalized
// rows into workspace. One block (256 thr) per row. Inputs are read exactly
// once -> non-temporal loads keep L2 free for the bf16 workspace the GEMM
// re-reads 16x.
// ---------------------------------------------------------------------------
__global__ __launch_bounds__(256)
void normalize_pack_kernel(const float* __restrict__ cont,
                           const float* __restrict__ aud,
                           unsigned short* __restrict__ cb,
                           unsigned short* __restrict__ ab,
                           float* __restrict__ diag_part) {
    const int row = blockIdx.x;
    const int tid = threadIdx.x;
    const v4f* c4 = (const v4f*)(cont + (size_t)row * DIMK);
    const v4f* a4 = (const v4f*)(aud  + (size_t)row * DIMK);

    v4f cv[8], av[8];
    float sc = 0.f, sa = 0.f, dt = 0.f;
#pragma unroll
    for (int it = 0; it < 8; ++it) {
        int idx = tid + it * 256;
        v4f c = __builtin_nontemporal_load(&c4[idx]);
        v4f a = __builtin_nontemporal_load(&a4[idx]);
        cv[it] = c; av[it] = a;
        sc += c.x*c.x + c.y*c.y + c.z*c.z + c.w*c.w;
        sa += a.x*a.x + a.y*a.y + a.z*a.z + a.w*a.w;
        dt += c.x*a.x + c.y*a.y + c.z*a.z + c.w*a.w;
    }
#pragma unroll
    for (int off = 16; off > 0; off >>= 1) {
        sc += __shfl_xor(sc, off, 32);
        sa += __shfl_xor(sa, off, 32);
        dt += __shfl_xor(dt, off, 32);
    }
    __shared__ float rsc[8], rsa[8], rdt[8];
    const int wid = tid >> 5, lane = tid & 31;
    if (lane == 0) { rsc[wid] = sc; rsa[wid] = sa; rdt[wid] = dt; }
    __syncthreads();
    float tsc = 0.f, tsa = 0.f, tdt = 0.f;
#pragma unroll
    for (int w = 0; w < 8; ++w) { tsc += rsc[w]; tsa += rsa[w]; tdt += rdt[w]; }

    const float inc = 1.f / fmaxf(sqrtf(tsc), 1e-12f);
    const float ina = 1.f / fmaxf(sqrtf(tsa), 1e-12f);

    uint2* cbo = (uint2*)(cb + (size_t)row * DIMK);
    uint2* abo = (uint2*)(ab + (size_t)row * DIMK);
#pragma unroll
    for (int it = 0; it < 8; ++it) {
        int idx = tid + it * 256;
        v4f c = cv[it], a = av[it];
        uint2 cp, ap;
        cp.x = f32_to_bf16(c.x * inc) | (f32_to_bf16(c.y * inc) << 16);
        cp.y = f32_to_bf16(c.z * inc) | (f32_to_bf16(c.w * inc) << 16);
        ap.x = f32_to_bf16(a.x * ina) | (f32_to_bf16(a.y * ina) << 16);
        ap.y = f32_to_bf16(a.z * ina) | (f32_to_bf16(a.w * ina) << 16);
        cbo[idx] = cp; abo[idx] = ap;
    }

    if (tid == 0) {
        // exact fp32 diagonal: scores_ii^2 = max(2 - 2*sim_ii, 0)
        float sim = tdt * inc * ina;
        diag_part[row] = fmaxf(2.f - 2.f * sim, 0.f);
    }
}

// ---------------------------------------------------------------------------
// Kernel 2: 2048x2048 bf16 Gram matrix via v_wmma_f32_16x16x32_bf16, fused
// hinge-loss epilogue. Block tile 128x128, BK=64, double-buffered LDS staged
// with GLOBAL_LOAD_ASYNC_TO_LDS_B128 (ASYNCcnt pipeline). 8 waves: wave
// (wm 0..3, wn 0..1) owns a 32x64 patch = 2x4 WMMA tiles, 16 WMMA / iter.
// ---------------------------------------------------------------------------
__global__ __launch_bounds__(256)
void gram_loss_kernel(const unsigned short* __restrict__ cb,
                      const unsigned short* __restrict__ ab,
                      float* __restrict__ gemm_part) {
    __shared__ __align__(16) unsigned short lds_c[2][BM * LDSS];
    __shared__ __align__(16) unsigned short lds_a[2][BN * LDSS];
    __shared__ float red[8];

    const int tid    = threadIdx.x;
    const int wid    = tid >> 5;
    const int lane   = tid & 31;
    const int wm     = wid & 3;      // 0..3 -> 32-row slice
    const int wn     = wid >> 2;     // 0..1 -> 64-col slice
    const int m_lane = lane & 15;
    const int half   = lane >> 4;

    const int bm = blockIdx.x * BM;
    const int bn = blockIdx.y * BN;

    const v8f zero = {0.f,0.f,0.f,0.f,0.f,0.f,0.f,0.f};
    v8f accf[2][4];
#pragma unroll
    for (int i = 0; i < 2; ++i)
#pragma unroll
        for (int j = 0; j < 4; ++j) accf[i][j] = zero;

    union FragBF { v16bf v; uint4 q[2]; };

    // Stage one 128x64 bf16 tile of each matrix (16 KiB each) into buffer `buf`.
    // 2048 16B chunks total; 8 per thread (4 per matrix).
    auto stage = [&](int t, int buf) {
        const int k0 = t * BK;
#pragma unroll
        for (int s = 0; s < 4; ++s) {
            int q  = tid + s * 256;      // 0..1023
            int r  = q >> 3;             // row 0..127
            int c8 = q & 7;              // 16B chunk within the 64-wide k slab
            const unsigned short* gc = cb + (size_t)(bm + r) * DIMK + k0 + c8 * 8;
            const unsigned short* ga = ab + (size_t)(bn + r) * DIMK + k0 + c8 * 8;
            unsigned short* lc = &lds_c[buf][r * LDSS + c8 * 8];
            unsigned short* la = &lds_a[buf][r * LDSS + c8 * 8];
#if HAVE_ASYNC_LDS
            async_cp_b128(gc, lc);
            async_cp_b128(ga, la);
#else
            *(uint4*)lc = *(const uint4*)gc;
            *(uint4*)la = *(const uint4*)ga;
#endif
        }
    };

    stage(0, 0);

    for (int t = 0; t < KTILES; ++t) {
#if HAVE_ASYNC_LDS
        wait_async0();                  // own tile-t asyncs landed in LDS
#endif
        __syncthreads();                // everyone's tile-t visible; t-1 consumers done
        if (t + 1 < KTILES) stage(t + 1, (t + 1) & 1);   // overlaps with consume below

        const unsigned short* bufc = lds_c[t & 1];
        const unsigned short* bufa = lds_a[t & 1];
#pragma unroll
        for (int ks = 0; ks < 2; ++ks) {
            // A frag (16x32): lane holds row m_lane; K = half*8+[0..7], 16+half*8+[0..7]
            FragBF afr[2];
#pragma unroll
            for (int i = 0; i < 2; ++i) {
                const unsigned short* base =
                    &bufc[(wm * 32 + i * 16 + m_lane) * LDSS + ks * 32];
                afr[i].q[0] = *(const uint4*)(base + half * 8);
                afr[i].q[1] = *(const uint4*)(base + 16 + half * 8);
            }
            // B frag (32x16): lane holds col m_lane; K = half*8+[0..7], 16+half*8+[0..7]
            FragBF bfr[4];
#pragma unroll
            for (int j = 0; j < 4; ++j) {
                const unsigned short* base =
                    &bufa[(wn * 64 + j * 16 + m_lane) * LDSS + ks * 32];
                bfr[j].q[0] = *(const uint4*)(base + half * 8);
                bfr[j].q[1] = *(const uint4*)(base + 16 + half * 8);
            }
#pragma unroll
            for (int i = 0; i < 2; ++i)
#pragma unroll
                for (int j = 0; j < 4; ++j)
                    accf[i][j] = __builtin_amdgcn_wmma_f32_16x16x32_bf16(
                        false, afr[i].v, false, bfr[j].v,
                        (short)0, accf[i][j], false, false);
        }
    }

    // Fused epilogue: scores = sqrt(max(2-2s,0)); off-diag hinge^2.
    // C/D layout: elem e -> row = half*8 + e, col = m_lane (within 16x16 tile).
    float lsum = 0.f;
#pragma unroll
    for (int i = 0; i < 2; ++i) {
#pragma unroll
        for (int j = 0; j < 4; ++j) {
            int r0 = bm + wm * 32 + i * 16 + half * 8;
            int c0 = bn + wn * 64 + j * 16 + m_lane;
#pragma unroll
            for (int e = 0; e < 8; ++e) {
                float s    = accf[i][j][e];
                float d2   = fmaxf(2.f - 2.f * s, 0.f);
                float dist = sqrtf(d2);
                float cost = fmaxf(MARGINF - dist, 0.f);
                lsum += ((r0 + e) == c0) ? 0.f : cost * cost;
            }
        }
    }
#pragma unroll
    for (int off = 16; off > 0; off >>= 1) lsum += __shfl_xor(lsum, off, 32);
    if (lane == 0) red[wid] = lsum;
    __syncthreads();
    if (tid == 0) {
        float t = 0.f;
#pragma unroll
        for (int w = 0; w < 8; ++w) t += red[w];
        gemm_part[blockIdx.y * gridDim.x + blockIdx.x] = t;
    }
}

// ---------------------------------------------------------------------------
// Kernel 3: deterministic final reduction of 2048 diag + 256 gemm partials.
// ---------------------------------------------------------------------------
__global__ __launch_bounds__(256)
void finalize_kernel(const float* __restrict__ diag_part,
                     const float* __restrict__ gemm_part,
                     float* __restrict__ out) {
    const int tid = threadIdx.x;
    float s = gemm_part[tid];
#pragma unroll
    for (int it = 0; it < N_ROWS / 256; ++it) s += diag_part[tid + it * 256];
#pragma unroll
    for (int off = 16; off > 0; off >>= 1) s += __shfl_xor(s, off, 32);
    __shared__ float red[8];
    if ((tid & 31) == 0) red[tid >> 5] = s;
    __syncthreads();
    if (tid == 0) {
        float t = 0.f;
#pragma unroll
        for (int w = 0; w < 8; ++w) t += red[w];
        out[0] = t * (1.0f / (2.0f * (float)N_ROWS));
    }
}

// ---------------------------------------------------------------------------
extern "C" void kernel_launch(void* const* d_in, const int* in_sizes, int n_in,
                              void* d_out, int out_size, void* d_ws, size_t ws_size,
                              hipStream_t stream) {
    const float* cont = (const float*)d_in[0];
    const float* aud  = (const float*)d_in[1];
    float* out = (float*)d_out;

    // Workspace layout:
    //   [0, 8K)        : diag partials (2048 floats)
    //   [8K, 9K)       : gemm block partials (256 floats)
    //   [16K, +32M)    : bf16 normalized content (2048*8192*2)
    //   [.., +32M)     : bf16 normalized audio
    char* ws = (char*)d_ws;
    float* diag_part = (float*)ws;
    float* gemm_part = (float*)(ws + 8192);
    unsigned short* cb = (unsigned short*)(ws + 16384);
    unsigned short* ab = (unsigned short*)(ws + 16384 + (size_t)N_ROWS * DIMK * 2);

    normalize_pack_kernel<<<N_ROWS, 256, 0, stream>>>(cont, aud, cb, ab, diag_part);

    dim3 grid(N_ROWS / BM, N_ROWS / BN);
    gram_loss_kernel<<<grid, 256, 0, stream>>>(cb, ab, gemm_part);

    finalize_kernel<<<1, 256, 0, stream>>>(diag_part, gemm_part, out);
}
